// Top1MoE_60997125538168
// MI455X (gfx1250) — matrix operand order repaired and checked by reading, not compiled, not run on hardware
//
#include <hip/hip_runtime.h>

#define TOKENS 8192
#define DDIM   1024
#define HDIM   2048
#define NEXP   8
#define KT1    (DDIM / 32)   // 32 k-tiles for X@W1
#define KT2    (HDIM / 32)   // 64 k-tiles for H@W2
#define NT1    (HDIM / 16)   // 128 n-tiles of W1
#define NT2    (DDIM / 16)   // 64 n-tiles of W2

typedef __attribute__((ext_vector_type(16))) __bf16 v16bf;
typedef __attribute__((ext_vector_type(8)))  float  v8f;

// ---------- helpers ----------
__device__ __forceinline__ __bf16 f2bf(float f) {
  unsigned u = __builtin_bit_cast(unsigned, f);
  unsigned r = (u + 0x7FFFu + ((u >> 16) & 1u)) >> 16;  // round-nearest-even
  unsigned short h = (unsigned short)r;
  return __builtin_bit_cast(__bf16, h);
}

// Element offset inside an A-fragment-major buffer laid out as
// [ktile][half][lane(32)][elem(16)] for a (rows x K) bf16 activation tile.
// Inverse of the ISA 16-bit A-matrix 16x32 VGPR layout.
__device__ __forceinline__ int a_pack_idx(int row, int k) {
  int kt  = k >> 5, ko = k & 31;
  int kg  = (ko >> 3) & 1;                         // lane group
  int v   = ((ko & 7) >> 1) + ((ko >> 4) << 2);    // VGPR index 0..7
  int e2  = 2 * v + (ko & 1);                      // element 0..15
  int ln  = (row & 15) + (kg << 4);
  return ((((kt << 1) + (row >> 4)) << 9)) + (ln << 4) + e2;  // 512 el per (kt,half)
}

// ---------- kernel 0: zero per-expert counters ----------
__global__ void moe_zero(int* __restrict__ cnt) {
  if (threadIdx.x < NEXP) cnt[threadIdx.x] = 0;
}

// ---------- kernel 1: gating (one wave32 per token) ----------
__global__ void moe_gate(const float* __restrict__ x, const float* __restrict__ gw,
                         const float* __restrict__ gb, int* __restrict__ cnt,
                         int* __restrict__ tlist, float* __restrict__ coef) {
  const int wave = threadIdx.x >> 5, lane = threadIdx.x & 31;
  const int tok  = blockIdx.x * 8 + wave;
  const float* xr = x + (size_t)tok * DDIM;

  float acc[NEXP];
#pragma unroll
  for (int e = 0; e < NEXP; ++e) acc[e] = 0.f;

  for (int k = lane; k < DDIM; k += 32) {
    float xv = xr[k];
    const float* g = gw + (size_t)k * NEXP;
#pragma unroll
    for (int e = 0; e < NEXP; ++e) acc[e] += xv * g[e];
  }
#pragma unroll
  for (int e = 0; e < NEXP; ++e) {
    float v = acc[e];
#pragma unroll
    for (int off = 16; off > 0; off >>= 1) v += __shfl_xor(v, off, 32);
    acc[e] = v + gb[e];
  }
  if (lane == 0) {
    int best = 0; float m = acc[0];
#pragma unroll
    for (int e = 1; e < NEXP; ++e) if (acc[e] > m) { m = acc[e]; best = e; }
    float s = 0.f;
#pragma unroll
    for (int e = 0; e < NEXP; ++e) s += __expf(acc[e] - m);
    int slot = atomicAdd(&cnt[best], 1);
    tlist[best * TOKENS + slot] = tok;
    coef[tok] = 1.f / s;
  }
}

// ---------- kernel 2: pack f32 KxN weights (per-expert stacked) into
// WMMA B-fragment-major bf16: [e][ntile][ktile][lane(32)][elem(16)].
__global__ void moe_pack_b(const float* __restrict__ w, __bf16* __restrict__ wp,
                           int K, int N) {
  const int gwave = (int)((blockIdx.x * blockDim.x + threadIdx.x) >> 5);
  const int lane  = threadIdx.x & 31;
  const int ktiles = K >> 5, ntiles = N >> 4;
  const int tilesPerExpert = ktiles * ntiles;
  const int e  = gwave / tilesPerExpert;
  const int t  = gwave % tilesPerExpert;
  const int nt = t / ktiles;
  const int kt = t % ktiles;
  const float* we = w + (size_t)e * K * N;
  const int col  = nt * 16 + (lane & 15);
  const int krow = kt * 32 + (lane >> 4) * 8;
  v16bf frag;
#pragma unroll
  for (int i = 0; i < 16; ++i)
    frag[i] = f2bf(we[(size_t)(krow + (i & 7) + ((i >> 3) << 4)) * N + col]);
  *(v16bf*)(wp + ((size_t)gwave << 9) + (lane << 4)) = frag;
}

// ---------- kernel 3: fused per-expert MLP, 32-token tiles, WMMA bf16 ----------
__global__ void __launch_bounds__(256, 1)
moe_expert(const float* __restrict__ x,
           const __bf16* __restrict__ w1p, const float* __restrict__ b1,
           const __bf16* __restrict__ w2p, const float* __restrict__ b2,
           const int* __restrict__ cnt, const int* __restrict__ tlist,
           const float* __restrict__ coef, float* __restrict__ out) {
  extern __shared__ char smem[];
  int*    toks = (int*)smem;                         // 32 ints
  float*  cfs  = (float*)(smem + 128);               // 32 floats
  __bf16* Xs   = (__bf16*)(smem + 256);              // packed 32x1024 (64 KB)
  __bf16* Hs   = (__bf16*)(smem + 256 + 65536);      // packed 32x2048 (128 KB)

  const int e    = blockIdx.y;
  const int tile = blockIdx.x;
  const int n    = cnt[e];
  if (tile * 32 >= n) return;          // uniform: EXEC stays full for WMMA

  const int tid  = threadIdx.x;
  const int wave = tid >> 5, lane = tid & 31;
  const int l15  = lane & 15, kgrp = lane >> 4;

  if (tid < 32) {
    int idx = tile * 32 + tid;
    if (idx >= n) idx = tile * 32;     // pad ragged tile with its first token
    int t = tlist[e * TOKENS + idx];
    toks[tid] = t;
    cfs[tid]  = coef[t];
  }
  __syncthreads();

  // Stage x tile (32 x 1024) into A-fragment-major bf16 LDS.
  for (int i = tid; i < 32 * DDIM / 4; i += 256) {
    int r  = i >> 8;
    int c4 = (i & 255) * 4;
    const float4 v = *(const float4*)(x + (size_t)toks[r] * DDIM + c4);
    Xs[a_pack_idx(r, c4 + 0)] = f2bf(v.x);
    Xs[a_pack_idx(r, c4 + 1)] = f2bf(v.y);
    Xs[a_pack_idx(r, c4 + 2)] = f2bf(v.z);
    Xs[a_pack_idx(r, c4 + 3)] = f2bf(v.w);
  }
  __syncthreads();

  const __bf16* w1e = w1p + (size_t)e * NT1 * KT1 * 512;
  const float*  b1e = b1 + (size_t)e * HDIM;

  // -------- Phase A: H = relu(X @ W1 + b1) -> packed bf16 LDS --------
  // 8 waves x 128 cols per pass; 2 passes cover H = 2048. 32 rows per wave.
#pragma unroll 1
  for (int p = 0; p < 2; ++p) {
    const int colBase = p * 1024 + wave * 128;
    const int ntBase  = colBase >> 4;
    v8f acc[2][8];
#pragma unroll
    for (int h = 0; h < 2; ++h)
#pragma unroll
      for (int nt = 0; nt < 8; ++nt) acc[h][nt] = 0.f;

#pragma unroll 1
    for (int kt = 0; kt < KT1; ++kt) {
      v16bf a0 = *(const v16bf*)(Xs + (((kt << 1) + 0) << 9) + (lane << 4));
      v16bf a1 = *(const v16bf*)(Xs + (((kt << 1) + 1) << 9) + (lane << 4));
      // speculative prefetch of next k-slab (unconditional: no loop branch)
      __builtin_prefetch(w1e + ((((size_t)ntBase) * KT1 + kt + 1) << 9) + (lane << 4), 0, 1);
      // issue all 8 fragment loads first so they clause and overlap the WMMAs
      v16bf barr[8];
#pragma unroll
      for (int nt = 0; nt < 8; ++nt)
        barr[nt] = *(const v16bf*)(w1e + ((((size_t)(ntBase + nt)) * KT1 + kt) << 9) + (lane << 4));
#pragma unroll
      for (int nt = 0; nt < 8; ++nt) {
        acc[0][nt] = __builtin_amdgcn_wmma_f32_16x16x32_bf16(
            false, a0, false, barr[nt], (short)0, acc[0][nt], false, false);
        acc[1][nt] = __builtin_amdgcn_wmma_f32_16x16x32_bf16(
            false, a1, false, barr[nt], (short)0, acc[1][nt], false, false);
      }
    }
#pragma unroll
    for (int nt = 0; nt < 8; ++nt) {
      int   colg = colBase + nt * 16 + l15;
      float bias = b1e[colg];
#pragma unroll
      for (int h = 0; h < 2; ++h)
#pragma unroll
        for (int r = 0; r < 8; ++r) {
          int M = h * 16 + r + 8 * kgrp;
          float v = acc[h][nt][r] + bias;
          v = v > 0.f ? v : 0.f;
          Hs[a_pack_idx(M, colg)] = f2bf(v);
        }
    }
  }
  __syncthreads();

  // -------- Phase B: OUT = coef * (H @ W2 + b2), scatter rows --------
  const __bf16* w2e = w2p + (size_t)e * NT2 * KT2 * 512;
  const float*  b2e = b2 + (size_t)e * DDIM;
  const int colBase = wave * 128;      // 8 waves x 128 = 1024 cols
  const int ntBase  = colBase >> 4;

  v8f acc[2][8];
#pragma unroll
  for (int h = 0; h < 2; ++h)
#pragma unroll
    for (int nt = 0; nt < 8; ++nt) acc[h][nt] = 0.f;

#pragma unroll 1
  for (int kt = 0; kt < KT2; ++kt) {
    v16bf a0 = *(const v16bf*)(Hs + (((kt << 1) + 0) << 9) + (lane << 4));
    v16bf a1 = *(const v16bf*)(Hs + (((kt << 1) + 1) << 9) + (lane << 4));
    __builtin_prefetch(w2e + ((((size_t)ntBase) * KT2 + kt + 1) << 9) + (lane << 4), 0, 1);
    v16bf barr[8];
#pragma unroll
    for (int nt = 0; nt < 8; ++nt)
      barr[nt] = *(const v16bf*)(w2e + ((((size_t)(ntBase + nt)) * KT2 + kt) << 9) + (lane << 4));
#pragma unroll
    for (int nt = 0; nt < 8; ++nt) {
      acc[0][nt] = __builtin_amdgcn_wmma_f32_16x16x32_bf16(
          false, a0, false, barr[nt], (short)0, acc[0][nt], false, false);
      acc[1][nt] = __builtin_amdgcn_wmma_f32_16x16x32_bf16(
          false, a1, false, barr[nt], (short)0, acc[1][nt], false, false);
    }
  }

  int valid = n - tile * 32; if (valid > 32) valid = 32;
#pragma unroll
  for (int nt = 0; nt < 8; ++nt) {
    int   colg = colBase + nt * 16 + l15;
    float bias = b2e[colg];
#pragma unroll
    for (int h = 0; h < 2; ++h)
#pragma unroll
      for (int r = 0; r < 8; ++r) {
        int M = h * 16 + r + 8 * kgrp;
        if (M < valid)
          out[(size_t)toks[M] * DDIM + colg] = cfs[M] * (acc[h][nt][r] + bias);
      }
  }
}

// ---------- launch ----------
extern "C" void kernel_launch(void* const* d_in, const int* in_sizes, int n_in,
                              void* d_out, int out_size, void* d_ws, size_t ws_size,
                              hipStream_t stream) {
  const float* x  = (const float*)d_in[0];
  const float* gw = (const float*)d_in[1];
  const float* gb = (const float*)d_in[2];
  const float* w1 = (const float*)d_in[3];
  const float* b1 = (const float*)d_in[4];
  const float* w2 = (const float*)d_in[5];
  const float* b2 = (const float*)d_in[6];
  float* out = (float*)d_out;

  char*   ws    = (char*)d_ws;
  int*    cnt   = (int*)ws;                                    // 8 ints
  int*    tlist = (int*)(ws + 256);                            // 8*8192 ints (256 KB)
  float*  coef  = (float*)(ws + 256 + (size_t)NEXP * TOKENS * sizeof(int));
  __bf16* w1p   = (__bf16*)(ws + 524288);                      // 32 MB packed W1
  __bf16* w2p   = (__bf16*)(ws + 524288 + (size_t)NEXP * NT1 * KT1 * 512 * 2);

  moe_zero<<<1, 32, 0, stream>>>(cnt);
  moe_gate<<<TOKENS / 8, 256, 0, stream>>>(x, gw, gb, cnt, tlist, coef);

  // pack weights: one wave per 32x16 tile
  int waves1 = NEXP * KT1 * NT1;   // 32768
  int waves2 = NEXP * KT2 * NT2;   // 32768
  moe_pack_b<<<waves1 * 32 / 256, 256, 0, stream>>>(w1, w1p, DDIM, HDIM);
  moe_pack_b<<<waves2 * 32 / 256, 256, 0, stream>>>(w2, w2p, HDIM, DDIM);

  size_t shmem = 256 + 65536 + 131072;   // 196,864 B (<= 320 KB WGP LDS)
  dim3 grid(TOKENS / 32, NEXP);
  moe_expert<<<grid, 256, shmem, stream>>>(x, w1p, b1, w2p, b2, cnt, tlist, coef, out);
}
